// Net_90194313216113
// MI455X (gfx1250) — matrix operand order repaired and checked by reading, not compiled, not run on hardware
//
#include <hip/hip_runtime.h>
#include <hip/hip_bf16.h>
#include <math.h>

// ---------------- types for WMMA ----------------
typedef __attribute__((ext_vector_type(16))) __bf16        v16bf;
typedef __attribute__((ext_vector_type(8)))  float         v8f;
typedef __attribute__((ext_vector_type(16))) unsigned short us16;

#define N_NODES 10000
#define K_IN    50176
#define N_HID   64
#define N_OUT   10
#define KCHUNKS (K_IN / 32)               // 1568
#define KSPLIT  8
#define CHUNKS_PER_WG (KCHUNKS / KSPLIT)  // 196
#define MTILE   64
#define XSTRIDE 36                        // f32 elems per LDS row (32 + 4 pad, bank-conflict free)

__device__ __forceinline__ unsigned short f2bf(float f) {
  // round-to-nearest-even f32 -> bf16 (bit-exact fallback; used off the hot path)
  unsigned u = __builtin_bit_cast(unsigned, f);
  u += 0x7FFFu + ((u >> 16) & 1u);
  return (unsigned short)(u >> 16);
}

__device__ __forceinline__ void atomAddF(float* p, float v) {
  __hip_atomic_fetch_add(p, v, __ATOMIC_RELAXED, __HIP_MEMORY_SCOPE_AGENT);
}

// ---------------- utility ----------------
__global__ void k_fill(float* __restrict__ p, float v, int n) {
  for (int i = blockIdx.x * blockDim.x + threadIdx.x; i < n; i += gridDim.x * blockDim.x)
    p[i] = v;
}

// ---------------- pack W1 (f32 row-major [K,64]) into bf16 B-fragment layout ----------------
// 16-bit B 32x16, contiguous k order (same as A in GEMM1): fragment for (kChunk kc, nTile nt):
//   lane l: n = nt*16 + (l&15), half = l>>4, elements j=0..15 are K = kc*32 + half*16 + j
// Stored contiguously: w1p[((kc*4 + nt)*32 + lane)*16 + j]
__global__ void k_pack_w1(const float* __restrict__ W1, unsigned short* __restrict__ w1p) {
  int t = blockIdx.x * blockDim.x + threadIdx.x;
  if (t >= KCHUNKS * 128) return;
  int kc   = t >> 7;
  int rem  = t & 127;
  int nt   = rem >> 5;
  int lane = rem & 31;
  int n    = nt * 16 + (lane & 15);
  int half = lane >> 4;
  us16 o;
#pragma unroll
  for (int j = 0; j < 16; ++j)
    o[j] = f2bf(W1[(size_t)(kc * 32 + half * 16 + j) * N_HID + n]);
  *(us16*)(w1p + (size_t)t * 16) = o;
}

// ---------------- GEMM1: H1 = X @ W1 ----------------
// X: double-buffered async-to-LDS streaming (ASYNCcnt). B: register double-buffer, one
// chunk ahead. bf16 WMMA, f32 accumulate; K split 8 ways with f32 atomic reduction.
__global__ void __launch_bounds__(128) k_gemm1(const float* __restrict__ x,
                                               const unsigned short* __restrict__ w1p,
                                               float* __restrict__ h1) {
  __shared__ __attribute__((aligned(16))) float ldsX[2][MTILE * XSTRIDE];

  const int tid   = threadIdx.x;
  const int lane  = tid & 31;
  const int wave  = tid >> 5;                 // 0..3, each owns 16 rows x 64 cols
  const int mBase = blockIdx.x * MTILE;
  const int ks    = blockIdx.y;               // K split index

  // X staging mapping: 2 threads per row, 16 f32 (64B = 4x b128 async) each.
  const int row  = tid >> 1;                  // 0..63
  const int seg  = tid & 1;                   // 0/1 -> cols 0..15 / 16..31
  int gRow = mBase + row;
  if (gRow >= N_NODES) gRow = 0;              // clamp: keeps EXEC all-ones, rows discarded at store
  const float* xRowPtr = x + (size_t)gRow * K_IN + seg * 16;
  const unsigned ldsRowOff[2] = {
    (unsigned)(size_t)&ldsX[0][row * XSTRIDE + seg * 16],
    (unsigned)(size_t)&ldsX[1][row * XSTRIDE + seg * 16]
  };

  // issue 64B of this thread's X slice for `chunk` into LDS buffer `buf`
  auto issue = [&](int buf, int chunk) {
    const float* gp = xRowPtr + (size_t)(ks * CHUNKS_PER_WG + chunk) * 32;
    unsigned la = ldsRowOff[buf];
    asm volatile(
        "global_load_async_to_lds_b128 %0, %1, off\n\t"
        "global_load_async_to_lds_b128 %0, %1, off offset:16\n\t"
        "global_load_async_to_lds_b128 %0, %1, off offset:32\n\t"
        "global_load_async_to_lds_b128 %0, %1, off offset:48"
        :: "v"(la), "v"(gp) : "memory");
  };

  auto loadB = [&](int chunk, v16bf& b0, v16bf& b1, v16bf& b2, v16bf& b3) {
    const int kc = ks * CHUNKS_PER_WG + chunk;
    const unsigned short* bptr = w1p + ((size_t)kc * 128 + lane) * 16;
    b0 = __builtin_bit_cast(v16bf, *(const us16*)(bptr + 0 * 512));
    b1 = __builtin_bit_cast(v16bf, *(const us16*)(bptr + 1 * 512));
    b2 = __builtin_bit_cast(v16bf, *(const us16*)(bptr + 2 * 512));
    b3 = __builtin_bit_cast(v16bf, *(const us16*)(bptr + 3 * 512));
    __builtin_prefetch(bptr + 2 * 2048, 0, 1);  // 2 chunks ahead (global_prefetch_b8)
  };

  // A fragment addressing (16-bit A 16x32): lane half selects K 0-15 vs 16-31 (matches packed B)
  const int mfrag = lane & 15;
  const int kbase = (lane >> 4) * 16;
  const float* lrowBase[2] = {
    &ldsX[0][(wave * 16 + mfrag) * XSTRIDE],
    &ldsX[1][(wave * 16 + mfrag) * XSTRIDE]
  };

  v8f acc0 = {}, acc1 = {}, acc2 = {}, acc3 = {};
  v16bf b0c, b1c, b2c, b3c, b0n, b1n, b2n, b3n;

  issue(0, 0);                                // preamble: chunk 0 -> LDS buffer 0
  loadB(0, b0c, b1c, b2c, b3c);               // preamble: chunk 0 B fragments

#pragma unroll 2
  for (int c = 0; c < CHUNKS_PER_WG; ++c) {
    const int cur = c & 1;
    // my async writes for buf[cur] landed; barrier => everyone's landed, and
    // everyone is done reading buf[cur^1] from the previous iteration.
    asm volatile("s_wait_asynccnt 0" ::: "memory");
    __syncthreads();
    const bool more = (c + 1 < CHUNKS_PER_WG);
    if (more) issue(cur ^ 1, c + 1);          // overlap next X chunk with compute

    // ---- A fragment: 16 f32 from LDS, native RNE convert to bf16 ----
    const float* lrow = lrowBase[cur];
    float4 p0 = *(const float4*)(lrow + kbase);
    float4 p1 = *(const float4*)(lrow + kbase + 4);
    float4 p2 = *(const float4*)(lrow + kbase + 8);
    float4 p3 = *(const float4*)(lrow + kbase + 12);
    v16bf a;
    a[0]  = (__bf16)p0.x; a[1]  = (__bf16)p0.y; a[2]  = (__bf16)p0.z; a[3]  = (__bf16)p0.w;
    a[4]  = (__bf16)p1.x; a[5]  = (__bf16)p1.y; a[6]  = (__bf16)p1.z; a[7]  = (__bf16)p1.w;
    a[8]  = (__bf16)p2.x; a[9]  = (__bf16)p2.y; a[10] = (__bf16)p2.z; a[11] = (__bf16)p2.w;
    a[12] = (__bf16)p3.x; a[13] = (__bf16)p3.y; a[14] = (__bf16)p3.z; a[15] = (__bf16)p3.w;

    // ---- pipeline next chunk's B fragments (consumed next iteration) ----
    if (more) loadB(c + 1, b0n, b1n, b2n, b3n);

    acc0 = __builtin_amdgcn_wmma_f32_16x16x32_bf16(false, a, false, b0c, (short)0, acc0, false, false);
    acc1 = __builtin_amdgcn_wmma_f32_16x16x32_bf16(false, a, false, b1c, (short)0, acc1, false, false);
    acc2 = __builtin_amdgcn_wmma_f32_16x16x32_bf16(false, a, false, b2c, (short)0, acc2, false, false);
    acc3 = __builtin_amdgcn_wmma_f32_16x16x32_bf16(false, a, false, b3c, (short)0, acc3, false, false);

    b0c = b0n; b1c = b1n; b2c = b2n; b3c = b3n;  // rotate (free after unroll renaming)
  }

  // ---- reduce K splits via f32 global atomics (C layout: lane half -> M {r, 8+r}) ----
  const int rBase = mBase + wave * 16 + (lane >> 4) * 8;
  const int col   = lane & 15;
#pragma unroll
  for (int r = 0; r < 8; ++r) {
    int gr = rBase + r;
    if (gr < N_NODES) {
      float* dst = h1 + (size_t)gr * N_HID + col;
      atomAddF(dst +  0, acc0[r]);
      atomAddF(dst + 16, acc1[r]);
      atomAddF(dst + 32, acc2[r]);
      atomAddF(dst + 48, acc3[r]);
    }
  }
}

// ---------------- degree / normalization ----------------
__global__ void k_deg(const long long* __restrict__ ei, float* __restrict__ deg, int E) {
  for (int e = blockIdx.x * blockDim.x + threadIdx.x; e < E; e += gridDim.x * blockDim.x)
    atomAddF(&deg[(int)ei[E + e]], 1.0f);
}

__global__ void k_rsqrt(float* __restrict__ p, int n) {
  int i = blockIdx.x * blockDim.x + threadIdx.x;
  if (i < n) p[i] = rsqrtf(p[i]);   // deg >= 1 always (self loop)
}

// ---------------- layer-1 aggregation ----------------
__global__ void k_self1(const float* __restrict__ h1, const float* __restrict__ dinv,
                        float* __restrict__ a1) {
  int t = blockIdx.x * blockDim.x + threadIdx.x;
  if (t >= N_NODES * N_HID) return;
  float s = dinv[t >> 6];
  a1[t] = h1[t] * s * s;
}

__global__ void k_edge1(const long long* __restrict__ ei, const float* __restrict__ h1,
                        const float* __restrict__ dinv, float* __restrict__ a1, int E) {
  int t = blockIdx.x * blockDim.x + threadIdx.x;
  if (t >= E * N_HID) return;
  int e = t >> 6, c = t & 63;
  int s = (int)ei[e], d = (int)ei[E + e];
  float nrm = dinv[s] * dinv[d];
  atomAddF(&a1[(size_t)d * N_HID + c], h1[(size_t)s * N_HID + c] * nrm);
}

__global__ void k_bias_relu(float* __restrict__ a1, const float* __restrict__ b1) {
  int t = blockIdx.x * blockDim.x + threadIdx.x;
  if (t >= N_NODES * N_HID) return;
  a1[t] = fmaxf(a1[t] + b1[t & 63], 0.0f);
}

// ---------------- tiny second GEMM: H2 = A1 @ W2  ([10000,64]@[64,10]) ----------------
__global__ void k_gemm2(const float* __restrict__ a1, const float* __restrict__ W2,
                        float* __restrict__ h2) {
  int t = blockIdx.x * blockDim.x + threadIdx.x;
  if (t >= N_NODES * N_OUT) return;
  int i = t / N_OUT, j = t % N_OUT;
  const float* row = a1 + (size_t)i * N_HID;
  float s = 0.0f;
#pragma unroll 8
  for (int c = 0; c < N_HID; ++c) s += row[c] * W2[c * N_OUT + j];
  h2[t] = s;
}

// ---------------- layer-2 aggregation (into d_out) ----------------
__global__ void k_self2(const float* __restrict__ h2, const float* __restrict__ dinv,
                        float* __restrict__ out) {
  int t = blockIdx.x * blockDim.x + threadIdx.x;
  if (t >= N_NODES * N_OUT) return;
  float s = dinv[t / N_OUT];
  out[t] = h2[t] * s * s;
}

__global__ void k_edge2(const long long* __restrict__ ei, const float* __restrict__ h2,
                        const float* __restrict__ dinv, float* __restrict__ out, int E) {
  int t = blockIdx.x * blockDim.x + threadIdx.x;
  if (t >= E * N_OUT) return;
  int e = t / N_OUT, j = t % N_OUT;
  int s = (int)ei[e], d = (int)ei[E + e];
  float nrm = dinv[s] * dinv[d];
  atomAddF(&out[d * N_OUT + j], h2[s * N_OUT + j] * nrm);
}

__global__ void k_final(float* __restrict__ out, const float* __restrict__ b2, int n) {
  int i = blockIdx.x * blockDim.x + threadIdx.x;
  if (i >= n) return;
  float v[N_OUT], m = -INFINITY;
#pragma unroll
  for (int j = 0; j < N_OUT; ++j) { v[j] = out[i * N_OUT + j] + b2[j]; m = fmaxf(m, v[j]); }
  float s = 0.0f;
#pragma unroll
  for (int j = 0; j < N_OUT; ++j) s += expf(v[j] - m);
  float l = logf(s);
#pragma unroll
  for (int j = 0; j < N_OUT; ++j) out[i * N_OUT + j] = v[j] - m - l;
}

// ---------------- launch ----------------
extern "C" void kernel_launch(void* const* d_in, const int* in_sizes, int n_in,
                              void* d_out, int out_size, void* d_ws, size_t ws_size,
                              hipStream_t stream) {
  const float*     x  = (const float*)d_in[0];
  const long long* ei = (const long long*)d_in[1];
  const float*     W1 = (const float*)d_in[2];
  const float*     b1 = (const float*)d_in[3];
  const float*     W2 = (const float*)d_in[4];
  const float*     b2 = (const float*)d_in[5];
  float* out = (float*)d_out;
  const int E = in_sizes[1] / 2;

  char* ws = (char*)d_ws;
  size_t off = 0;
  auto take = [&](size_t bytes) -> void* {
    void* p = ws + off;
    off = (off + bytes + 255) & ~(size_t)255;
    return p;
  };
  unsigned short* w1p  = (unsigned short*)take((size_t)KCHUNKS * 128 * 16 * 2); // 6.4 MB
  float*          h1   = (float*)take((size_t)N_NODES * N_HID * 4);
  float*          a1   = (float*)take((size_t)N_NODES * N_HID * 4);
  float*          dinv = (float*)take((size_t)N_NODES * 4);
  float*          h2   = (float*)take((size_t)N_NODES * N_OUT * 4);

  const int B = 256;
  // pack W1 into bf16 fragment layout (L2-resident for GEMM1)
  k_pack_w1<<<(KCHUNKS * 128 + B - 1) / B, B, 0, stream>>>(W1, w1p);
  // H1 = 0, then K-split WMMA GEMM with atomic reduction
  k_fill<<<(N_NODES * N_HID + B - 1) / B, B, 0, stream>>>(h1, 0.0f, N_NODES * N_HID);
  dim3 g1((N_NODES + MTILE - 1) / MTILE, KSPLIT);
  k_gemm1<<<g1, 128, 0, stream>>>(x, w1p, h1);
  // normalization
  k_fill<<<(N_NODES + B - 1) / B, B, 0, stream>>>(dinv, 1.0f, N_NODES);
  k_deg<<<(E + B - 1) / B, B, 0, stream>>>(ei, dinv, E);
  k_rsqrt<<<(N_NODES + B - 1) / B, B, 0, stream>>>(dinv, N_NODES);
  // layer-1 aggregate + bias + relu
  k_self1<<<(N_NODES * N_HID + B - 1) / B, B, 0, stream>>>(h1, dinv, a1);
  k_edge1<<<(E * N_HID + B - 1) / B, B, 0, stream>>>(ei, h1, dinv, a1, E);
  k_bias_relu<<<(N_NODES * N_HID + B - 1) / B, B, 0, stream>>>(a1, b1);
  // layer 2
  k_gemm2<<<(N_NODES * N_OUT + B - 1) / B, B, 0, stream>>>(a1, W2, h2);
  k_self2<<<(N_NODES * N_OUT + B - 1) / B, B, 0, stream>>>(h2, dinv, out);
  k_edge2<<<(E * N_OUT + B - 1) / B, B, 0, stream>>>(ei, h2, dinv, out, E);
  k_final<<<(N_NODES + B - 1) / B, B, 0, stream>>>(out, b2, N_NODES);
}
